// GNN_72696616452564
// MI455X (gfx1250) — matrix operand order repaired and checked by reading, not compiled, not run on hardware
//
#include <hip/hip_runtime.h>

#define N_NODES 25000
#define E_TOT   400000
#define HID     16
#define MLPW    128
#define NTILES  (E_TOT / 128)   // 3125
#define EGRID   625             // 3125 = 625 * 5 tiles per persistent block

typedef __attribute__((ext_vector_type(16))) __bf16 v16bf;
typedef __attribute__((ext_vector_type(8)))  float  v8f;
typedef __attribute__((ext_vector_type(4)))  unsigned int u32x4;
typedef __attribute__((ext_vector_type(8)))  unsigned int u32x8;

union FragB {
  v16bf v;
  unsigned short s[16];
  uint4 q[2];
};

__device__ __forceinline__ unsigned short f2bf(float f) {
  __bf16 b = (__bf16)f;                       // native v_cvt (RNE) on gfx1250
  return __builtin_bit_cast(unsigned short, b);
}

__device__ __forceinline__ v16bf load_frag16(const unsigned short* p) {
  FragB f;
  f.q[0] = *(const uint4*)p;
  f.q[1] = *(const uint4*)(p + 8);
  return f.v;
}

__device__ __forceinline__ v8f wmma_bf16(v16bf a, v16bf b, v8f c) {
  return __builtin_amdgcn_wmma_f32_16x16x32_bf16(false, a, false, b, (short)0, c,
                                                 false, false);
}
__device__ __forceinline__ v8f wmma_bf16_z(v16bf a, v16bf b) {
  v8f z = {0.f, 0.f, 0.f, 0.f, 0.f, 0.f, 0.f, 0.f};   // encodes SRC2 = inline 0
  return __builtin_amdgcn_wmma_f32_16x16x32_bf16(false, a, false, b, (short)0, z,
                                                 false, false);
}

// -------- weight prep: f32 [K,N] row-major -> bf16 WMMA B-fragment order ----
// fragment (kt,nt): 32 lanes x 16 elems; lane L: n = L%16, h = L/16;
// element e maps to K = kt*32 + 8*h + (e&7) + 16*(e>>3)  (ISA 16-bit B 32x16 layout).
// For the K-padded first layer, row Ksrc carries the bias (A supplies 1.0).
__global__ void wprep(const float* __restrict__ src, const float* __restrict__ bias,
                      unsigned short* __restrict__ dst, int Ksrc, int Nout, int total) {
  int idx = blockIdx.x * 256 + threadIdx.x;
  if (idx >= total) return;
  int frag = idx >> 9, wi = idx & 511;
  int lane = wi >> 4, e = wi & 15;
  int NTn = Nout >> 4;
  int kt = frag / NTn, nt = frag - kt * NTn;
  int row = kt * 32 + ((lane >> 4) << 3) + (e & 7) + ((e >> 3) << 4);
  int col = nt * 16 + (lane & 15);
  float v = 0.f;
  if (row < Ksrc)               v = src[row * Nout + col];
  else if (bias && row == Ksrc) v = bias[col];
  dst[idx] = f2bf(v);
}

__global__ void fillz(float* __restrict__ p, int n) {
  int i = blockIdx.x * 256 + threadIdx.x;
  if (i < n) p[i] = 0.f;
}

// -------- fused edge kernel: edge-MLP (WMMA) + message + scatter-add --------
template<int IN, int OUT>
__global__ __launch_bounds__(256, 1)
void edge_conv(const float* __restrict__ xin, const int* __restrict__ ei,
               const float* __restrict__ ea, const unsigned short* __restrict__ wf,
               const float* __restrict__ b1g, const float* __restrict__ b2g,
               const float* __restrict__ b3g, float* __restrict__ agg) {
  constexpr int NT   = MLPW / 16;            // 8 tiles across hidden width
  constexpr int NTO  = OUT / 16;             // 8 or 16
  constexpr int RS   = MLPW + 8;             // bank-padded act row stride (ushort)
  constexpr int WCNT = (9 * NT + 4 * NTO) * 512;   // bf16 weight elems in LDS
  constexpr int BCNT = 2 * MLPW + OUT;             // bias floats (b1,b2,b3)

  extern __shared__ unsigned char smem[];
  unsigned short* wlds  = (unsigned short*)smem;
  float*          blds  = (float*)(smem + (size_t)WCNT * 2);
  float*          xlds  = blds + BCNT;                    // [8][16][IN]
  int*            sdl   = (int*)(xlds + 8 * 16 * IN);     // [8][16][2] src,dst
  unsigned short* actall= (unsigned short*)(sdl + 8 * 16 * 2); // [8][16][RS]

  const int tid  = threadIdx.x;
  const int lane = tid & 31, w = tid >> 5;
  const int nM   = lane & 15, h = lane >> 4;

  // ---- weights -> LDS via Tensor Data Mover (1-D tile, 8B elements) -------
  if (w == 0) {
    const unsigned long long ga = (unsigned long long)(const void*)wf;
    const unsigned int n8 = (unsigned int)(WCNT * 2 / 8);
    u32x4 g0;
    g0[0] = 1u;                                   // count=1, user descriptor
    g0[1] = 0u;                                   // lds_addr = smem base
    g0[2] = (unsigned int)ga;                     // global_addr[31:0]
    g0[3] = (unsigned int)((ga >> 32) & 0x1FFFFFFu) | (2u << 30);  // [56:32]+type=2
    u32x8 g1;
    g1[0] = 3u << 16;                             // data_size = 8B
    g1[1] = (n8 & 0xFFFFu) << 16;                 // tensor_dim0[15:0]
    g1[2] = (n8 >> 16) | (1u << 16);              // tensor_dim0[31:16], tensor_dim1=1
    g1[3] = (n8 & 0xFFFFu) << 16;                 // tile_dim0
    g1[4] = 1u;                                   // tile_dim1=1, tile_dim2=0
    g1[5] = n8;                                   // tensor_dim0_stride[31:0]
    g1[6] = 0u;
    g1[7] = 0u;
    asm volatile("tensor_load_to_lds %0, %1" :: "s"(g0), "s"(g1) : "memory");
    __builtin_amdgcn_s_wait_tensorcnt(0);
  }
  // safety overwrite with identical bytes (descriptor not runtime-verifiable)
  for (int i = tid; i < WCNT; i += 256) wlds[i] = wf[i];
  for (int i = tid; i < 2 * MLPW; i += 256)
    blds[i] = (i < MLPW) ? b1g[i] : b2g[i - MLPW];
  for (int i = tid; i < OUT; i += 256) blds[2 * MLPW + i] = b3g[i];
  __syncthreads();

  unsigned short* act = actall + w * (16 * RS);

  for (int tile = blockIdx.x; tile < NTILES; tile += gridDim.x) {
    const int ebase = tile * 128 + w * 16;
    if (h == 0) {
      const int e = ebase + nM;
      const int s = ei[e];
      sdl[(w * 16 + nM) * 2 + 0] = s;
      sdl[(w * 16 + nM) * 2 + 1] = ei[E_TOT + e];
      #pragma unroll
      for (int i = 0; i < IN; ++i) xlds[(w * 16 + nM) * IN + i] = xin[(size_t)s * IN + i];
    }

    // layer-0 A fragment from edge_attr (K padded 4->32; K=4 element = 1.0
    // so the bias row folded into W0 is applied by the WMMA itself).
    // A 16x32 layout: lane M=L%16, h=L/16; elems 0..7 <-> K=8h..8h+7
    FragB a0;
    #pragma unroll
    for (int e = 0; e < 16; ++e) a0.s[e] = 0;
    if (h == 0) {
      const float4 v = *(const float4*)(ea + (size_t)(ebase + nM) * 4);
      a0.s[0] = f2bf(v.x); a0.s[1] = f2bf(v.y); a0.s[2] = f2bf(v.z); a0.s[3] = f2bf(v.w);
      a0.s[4] = 0x3F80;   // 1.0 (bf16) -> multiplies bias row of W0
    }
    __syncthreads();

    v8f acc[NT];
    // ---- MLP layer 0: [16e,32] x [32,128], C = inline 0, bias in weights
    #pragma unroll
    for (int nt = 0; nt < NT; ++nt)
      acc[nt] = wmma_bf16_z(a0.v, load_frag16(wlds + nt * 512 + lane * 16));
    #pragma unroll
    for (int nt = 0; nt < NT; ++nt)
      #pragma unroll
      for (int r = 0; r < 8; ++r)
        act[(r + 8 * h) * RS + nt * 16 + nM] = f2bf(fmaxf(acc[nt][r], 0.f));
    __syncthreads();

    // ---- MLP hidden layers 1,2: [16e,128] x [128,128]; bias added at store
    for (int layer = 0; layer < 2; ++layer) {
      const unsigned short* wh = wlds + (NT + layer * 4 * NT) * 512;
      const float* bl = blds + layer * MLPW;
      #pragma unroll
      for (int kt = 0; kt < 4; ++kt) {
        const unsigned short* ar = act + nM * RS + kt * 32 + 8 * h;
        FragB fa;
        fa.q[0] = *(const uint4*)ar;          // K = kt*32+8h .. +7
        fa.q[1] = *(const uint4*)(ar + 16);   // K = kt*32+16+8h .. +7
        #pragma unroll
        for (int nt = 0; nt < NT; ++nt) {
          const v16bf bfr = load_frag16(wh + (kt * NT + nt) * 512 + lane * 16);
          acc[nt] = (kt == 0) ? wmma_bf16_z(fa.v, bfr) : wmma_bf16(fa.v, bfr, acc[nt]);
        }
      }
      __syncthreads();
      #pragma unroll
      for (int nt = 0; nt < NT; ++nt) {
        const float bv = bl[nt * 16 + nM];
        #pragma unroll
        for (int r = 0; r < 8; ++r)
          act[(r + 8 * h) * RS + nt * 16 + nM] = f2bf(fmaxf(acc[nt][r] + bv, 0.f));
      }
      __syncthreads();
    }

    // ---- head: We = h3 @ W3 + b3; output tile nt == input channel i, so
    // msg[e][j] = sum_i x[src][i] * We[e][i*16+j] is computed in registers
    // from the D fragments (lane holds col j = nM, rows = edges r+8h).
    const unsigned short* w3 = wlds + (9 * NT) * 512;
    float msg[8];
    #pragma unroll
    for (int r = 0; r < 8; ++r) msg[r] = 0.f;

    #pragma unroll
    for (int ntb = 0; ntb < NTO; ntb += 8) {
      v8f wacc[8];
      #pragma unroll
      for (int j = 0; j < 8; ++j) {
        const float bv = blds[2 * MLPW + (ntb + j) * 16 + nM];
        #pragma unroll
        for (int r = 0; r < 8; ++r) wacc[j][r] = bv;
      }
      #pragma unroll
      for (int kt = 0; kt < 4; ++kt) {
        const unsigned short* ar = act + nM * RS + kt * 32 + 8 * h;
        FragB fa;
        fa.q[0] = *(const uint4*)ar;
        fa.q[1] = *(const uint4*)(ar + 16);
        #pragma unroll
        for (int j = 0; j < 8; ++j)
          wacc[j] = wmma_bf16(fa.v, load_frag16(w3 + (kt * NTO + ntb + j) * 512 + lane * 16),
                              wacc[j]);
      }
      #pragma unroll
      for (int r = 0; r < 8; ++r) {
        const int er = r + 8 * h;
        #pragma unroll
        for (int j = 0; j < 8; ++j)
          msg[r] += xlds[(w * 16 + er) * IN + (ntb + j)] * wacc[j][r];
      }
    }
    #pragma unroll
    for (int r = 0; r < 8; ++r) {
      const int er = r + 8 * h;
      const int d = sdl[(w * 16 + er) * 2 + 1];
      unsafeAtomicAdd(&agg[(size_t)d * HID + nM], msg[r]);
    }
    __syncthreads();   // xlds/sdl reused by next tile
  }
}

// -------- node update: z = agg + x@root + bias; accumulate BN sums ----------
template<int IN>
__global__ void node_update(const float* __restrict__ xin, const float* __restrict__ root,
                            const float* __restrict__ bias, const float* __restrict__ agg,
                            float* __restrict__ z, double* __restrict__ sum,
                            double* __restrict__ sq, int Nn) {
  __shared__ float rt[IN * 16];
  __shared__ float bs[16];
  __shared__ float s1[256], s2[256];
  const int tid = threadIdx.x;
  for (int i = tid; i < IN * 16; i += 256) rt[i] = root[i];
  if (tid < 16) bs[tid] = bias[tid];
  __syncthreads();
  const int col = tid & 15, nl = tid >> 4;
  const int node = blockIdx.x * 16 + nl;
  float zv = 0.f;
  if (node < Nn) {
    zv = agg[(size_t)node * 16 + col] + bs[col];
    #pragma unroll
    for (int i = 0; i < IN; ++i) zv += xin[(size_t)node * IN + i] * rt[i * 16 + col];
    z[(size_t)node * 16 + col] = zv;
  }
  s1[tid] = zv; s2[tid] = zv * zv;
  __syncthreads();
  if (tid < 16) {
    float a = 0.f, b = 0.f;
    #pragma unroll
    for (int k = 0; k < 16; ++k) { a += s1[k * 16 + tid]; b += s2[k * 16 + tid]; }
    unsafeAtomicAdd(&sum[tid], (double)a);
    unsafeAtomicAdd(&sq[tid], (double)b);
  }
}

__global__ void bn_relu(const float* __restrict__ z, const double* __restrict__ sum,
                        const double* __restrict__ sq, const float* __restrict__ g,
                        const float* __restrict__ b, float* __restrict__ hout, int Nn) {
  __shared__ float sc[16], shf[16];
  const int tid = threadIdx.x;
  if (tid < 16) {
    double mu = sum[tid] / Nn;
    double var = sq[tid] / Nn - mu * mu;
    float r = rsqrtf((float)var + 1e-5f);
    sc[tid] = g[tid] * r;
    shf[tid] = b[tid] - (float)mu * g[tid] * r;
  }
  __syncthreads();
  int idx = blockIdx.x * 256 + tid;
  if (idx >= Nn * 16) return;
  int c = idx & 15;
  hout[idx] = fmaxf(z[idx] * sc[c] + shf[c], 0.f);
}

__global__ void final_fc(const float* __restrict__ z, const double* __restrict__ sum,
                         const double* __restrict__ sq, const float* __restrict__ g,
                         const float* __restrict__ b, const float* __restrict__ fw,
                         const float* __restrict__ fb, float* __restrict__ out, int Nn) {
  __shared__ float sc[16], shf[16], fwl[32], fbl[2];
  const int tid = threadIdx.x;
  if (tid < 16) {
    double mu = sum[tid] / Nn;
    double var = sq[tid] / Nn - mu * mu;
    float r = rsqrtf((float)var + 1e-5f);
    sc[tid] = g[tid] * r;
    shf[tid] = b[tid] - (float)mu * g[tid] * r;
  }
  if (tid < 32) fwl[tid] = fw[tid];
  if (tid < 2) fbl[tid] = fb[tid];
  __syncthreads();
  int node = blockIdx.x * 256 + tid;
  if (node >= Nn) return;
  float o0 = fbl[0], o1 = fbl[1];
  #pragma unroll
  for (int c = 0; c < 16; ++c) {
    float v = fmaxf(z[(size_t)node * 16 + c] * sc[c] + shf[c], 0.f);
    o0 += v * fwl[c * 2 + 0];
    o1 += v * fwl[c * 2 + 1];
  }
  out[node * 2 + 0] = o0;
  out[node * 2 + 1] = o1;
}

// ---------------------------- host orchestration ----------------------------
static constexpr size_t OFF_AGG1 = 245760;           // bytes (after bf16 weights)
static constexpr size_t OFF_AGG2 = OFF_AGG1 + (size_t)N_NODES * 16 * 4;
static constexpr size_t OFF_Z1   = OFF_AGG2 + (size_t)N_NODES * 16 * 4;
static constexpr size_t OFF_H1   = OFF_Z1   + (size_t)N_NODES * 16 * 4;
static constexpr size_t OFF_Z2   = OFF_H1   + (size_t)N_NODES * 16 * 4;
static constexpr size_t OFF_SUMS = OFF_Z2   + (size_t)N_NODES * 16 * 4;

static constexpr int lds_bytes(int IN, int OUT) {
  int NT = MLPW / 16, NTO = OUT / 16;
  return (9 * NT + 4 * NTO) * 512 * 2      // weights (bf16)
       + (2 * MLPW + OUT) * 4              // biases b1,b2,b3
       + 8 * 16 * IN * 4                   // x[src] stage
       + 8 * 16 * 2 * 4                    // src/dst
       + 8 * 16 * (MLPW + 8) * 2;          // activation tiles
}

extern "C" void kernel_launch(void* const* d_in, const int* in_sizes, int n_in,
                              void* d_out, int out_size, void* d_ws, size_t ws_size,
                              hipStream_t stream) {
  const float* x   = (const float*)d_in[0];
  const int*   ei  = (const int*)d_in[1];
  const float* ea  = (const float*)d_in[2];
  const float* en1w[4] = {(const float*)d_in[3], (const float*)d_in[4], (const float*)d_in[5], (const float*)d_in[6]};
  const float* en1b[4] = {(const float*)d_in[7], (const float*)d_in[8], (const float*)d_in[9], (const float*)d_in[10]};
  const float* en2w[4] = {(const float*)d_in[11], (const float*)d_in[12], (const float*)d_in[13], (const float*)d_in[14]};
  const float* en2b[4] = {(const float*)d_in[15], (const float*)d_in[16], (const float*)d_in[17], (const float*)d_in[18]};
  const float* root1 = (const float*)d_in[19];
  const float* bias1 = (const float*)d_in[20];
  const float* root2 = (const float*)d_in[21];
  const float* bias2 = (const float*)d_in[22];
  const float* bn1g  = (const float*)d_in[23];
  const float* bn1b  = (const float*)d_in[24];
  const float* bn2g  = (const float*)d_in[25];
  const float* bn2b  = (const float*)d_in[26];
  const float* fcw   = (const float*)d_in[27];
  const float* fcb   = (const float*)d_in[28];

  unsigned char* ws = (unsigned char*)d_ws;
  unsigned short* wfrag = (unsigned short*)ws;
  float* agg1 = (float*)(ws + OFF_AGG1);
  float* agg2 = (float*)(ws + OFF_AGG2);
  float* z1   = (float*)(ws + OFF_Z1);
  float* h1   = (float*)(ws + OFF_H1);
  float* z2   = (float*)(ws + OFF_Z2);
  double* sums = (double*)(ws + OFF_SUMS);  // [sum1(16), sq1(16), sum2(16), sq2(16)]

  // 1) weight prep into fragment order (bf16); layer-0 bias folded into row 4
  struct { const float* s; const float* b; int Ksrc, KP, N, off; } mats[8] = {
    {en1w[0], en1b[0],   4,  32, 128,      0},
    {en1w[1], nullptr, 128, 128, 128,   4096},
    {en1w[2], nullptr, 128, 128, 128,  20480},
    {en1w[3], nullptr, 128, 128, 128,  36864},
    {en2w[0], en2b[0],   4,  32, 128,  53248},
    {en2w[1], nullptr, 128, 128, 128,  57344},
    {en2w[2], nullptr, 128, 128, 128,  73728},
    {en2w[3], nullptr, 128, 128, 256,  90112},
  };
  for (int m = 0; m < 8; ++m) {
    int total = (mats[m].KP / 32) * (mats[m].N / 16) * 512;
    wprep<<<(total + 255) / 256, 256, 0, stream>>>(mats[m].s, mats[m].b,
                                                   wfrag + mats[m].off,
                                                   mats[m].Ksrc, mats[m].N, total);
  }

  // 2) zero aggregation buffers + BN sums (every call: atomics accumulate)
  fillz<<<(2 * N_NODES * 16 + 255) / 256, 256, 0, stream>>>(agg1, 2 * N_NODES * 16);
  fillz<<<1, 256, 0, stream>>>((float*)sums, 128);

  // 3) layer 1: fused edge MLP + message + scatter (persistent blocks)
  edge_conv<8, 128><<<EGRID, 256, lds_bytes(8, 128), stream>>>(
      x, ei, ea, wfrag, en1b[1], en1b[2], en1b[3], agg1);

  // 4) node update + BN stats, then BN+ReLU
  node_update<8><<<(N_NODES + 15) / 16, 256, 0, stream>>>(
      x, root1, bias1, agg1, z1, sums + 0, sums + 16, N_NODES);
  bn_relu<<<(N_NODES * 16 + 255) / 256, 256, 0, stream>>>(
      z1, sums + 0, sums + 16, bn1g, bn1b, h1, N_NODES);

  // 5) layer 2
  edge_conv<16, 256><<<EGRID, 256, lds_bytes(16, 256), stream>>>(
      h1, ei, ea, wfrag + 53248, en2b[1], en2b[2], en2b[3], agg2);

  node_update<16><<<(N_NODES + 15) / 16, 256, 0, stream>>>(
      h1, root2, bias2, agg2, z2, sums + 32, sums + 48, N_NODES);

  // 6) BN + ReLU + final 16x2 FC
  final_fc<<<(N_NODES + 255) / 256, 256, 0, stream>>>(
      z2, sums + 32, sums + 48, bn2g, bn2b, fcw, fcb, (float*)d_out, N_NODES);
}
